// CausalAttention_44315472560263
// MI455X (gfx1250) — matrix operand order repaired and checked by reading, not compiled, not run on hardware
//
#include <hip/hip_runtime.h>
#include <stdint.h>

// ---------------------------------------------------------------------------
// Causal attention for MI455X (gfx1250, wave32, WMMA).
//   Phase 1: QKV = x @ [Wq|Wk|Wv]  -> bf16 Q,K (row major), V^T (d-major) in ws
//   Phase 2: flash attention with online softmax, f32 accumulation,
//            all matmuls via v_wmma_f32_16x16x32_bf16.
// Compute-bound (~174 GFLOP vs ~250MB HBM traffic): bf16 matrix path,
// async-to-LDS staging (ASYNCcnt), native bf16 converts.
// Workspace: 3 * 16384*1024 bf16 = ~100.7 MB.
// ---------------------------------------------------------------------------

typedef __attribute__((ext_vector_type(16))) __bf16 v16bf;
typedef __attribute__((ext_vector_type(2)))  __bf16 v2bf;
typedef __attribute__((ext_vector_type(8)))  float  v8f;
typedef __attribute__((ext_vector_type(4)))  int    v4i;

// pointer-to-(addrspace-qualified v4i) types for the async builtin
typedef __attribute__((address_space(1))) v4i* gv4i_p;   // global
typedef __attribute__((address_space(3))) v4i* lv4i_p;   // LDS

union FragBF {
  uint4 q[2];
  v16bf v;
};

#ifndef __HIP_DEVICE_COMPILE__
#define ASYNC_OK 0
#elif __has_builtin(__builtin_amdgcn_global_load_async_to_lds_b128)
#define ASYNC_OK 1
#else
#define ASYNC_OK 0
#warning "gfx1250 async-to-LDS builtin not available; using synchronous staging"
#endif

// Copy 16 bytes global -> LDS. Async variant uses the CDNA5 ASYNCcnt path
// (GLOBAL_LOAD_ASYNC_TO_LDS_B128): LDS dest address is per-lane, no VGPR data.
__device__ __forceinline__ void cp16(void* lds, const void* g) {
#if ASYNC_OK
  __builtin_amdgcn_global_load_async_to_lds_b128(
      (gv4i_p)(uintptr_t)g,
      (lv4i_p)(uint32_t)(uintptr_t)lds,   // low 32 bits of generic = LDS offset
      0, 0);
#else
  *(uint4*)lds = *(const uint4*)g;
#endif
}

// Wave-local completion of outstanding async copies (before the barrier).
__device__ __forceinline__ void async_join() {
#if ASYNC_OK
#if __has_builtin(__builtin_amdgcn_s_wait_asynccnt)
  __builtin_amdgcn_s_wait_asynccnt(0);
#else
  asm volatile("s_wait_asynccnt 0" ::: "memory");
#endif
#endif
}

// Native fptrunc f32->bf16 (RNE); lets isel pick v_cvt_pk_bf16_f32 if present.
__device__ __forceinline__ uint32_t f2bf_pk(float lo, float hi) {
  union { v2bf v; uint32_t u; } cv;
  cv.v[0] = (__bf16)lo;
  cv.v[1] = (__bf16)hi;
  return cv.u;
}

__device__ __forceinline__ uint16_t f2bf(float f) {
  union { __bf16 h; uint16_t u; } cv;
  cv.h = (__bf16)f;
  return cv.u;
}

#define D_MODEL 1024
#define SEQ     2048
#define NBATCH  8
#define MTOT    (NBATCH * SEQ)        // 16384 rows total

// ---------------------------------------------------------------------------
// Kernel 1: QKV projection GEMM.  [16384 x 1024] x [1024 x 3072]
// Block: 128 thr (4 waves), tile 128(M) x 128(N), K chunks of 32.
// Each wave: 64x64 -> 4x4 WMMA accumulators. fp32->bf16 conversion while
// staging, so staging stays on the synchronous path here.
// ---------------------------------------------------------------------------
__global__ __launch_bounds__(128)
void qkv_gemm_kernel(const float* __restrict__ x,
                     const float* __restrict__ Wq,
                     const float* __restrict__ Wk,
                     const float* __restrict__ Wv,
                     uint16_t* __restrict__ Qo,
                     uint16_t* __restrict__ Ko,
                     uint16_t* __restrict__ Vt) {
  __shared__ uint16_t sA[128 * 32];   // x tile, row major [m][k]
  __shared__ uint16_t sB[128 * 32];   // W tile, transposed [n][k]

  const int tid  = threadIdx.x;
  const int lane = tid & 31;
  const int wid  = tid >> 5;
  const int half = lane >> 4;         // 0/1 (wave halves)
  const int l16  = lane & 15;

  const int mt = blockIdx.x;          // 0..127
  const int nt = blockIdx.y;          // 0..23
  const int wsel = nt >> 3;           // 0=Q 1=K 2=V
  const int n0 = (nt & 7) * 128;      // column within selected weight
  const int m0 = mt * 128;

  const float* W = (wsel == 0) ? Wq : (wsel == 1) ? Wk : Wv;

  const int wr = wid >> 1;            // wave M offset = wr*64
  const int wc = wid & 1;             // wave N offset = wc*64

  v8f acc[4][4];
#pragma unroll
  for (int i = 0; i < 4; ++i)
#pragma unroll
    for (int j = 0; j < 4; ++j)
#pragma unroll
      for (int v = 0; v < 8; ++v) acc[i][j][v] = 0.0f;

  for (int kc = 0; kc < 32; ++kc) {
    const int kb = kc * 32;
    __syncthreads();
    // ---- stage A (fp32 -> bf16): thread = one row of 32 k-values ----
    {
      const float* src = x + (size_t)(m0 + tid) * D_MODEL + kb;
#pragma unroll
      for (int j = 0; j < 8; ++j) {
        float4 f = *(const float4*)(src + 4 * j);
        uint2 p;
        p.x = f2bf_pk(f.x, f.y);
        p.y = f2bf_pk(f.z, f.w);
        *(uint2*)&sA[tid * 32 + 4 * j] = p;
      }
    }
    // ---- stage B transposed: sB[n][k], packed k-pairs -> b32 stores ----
    {
      const int n4  = (tid & 31) * 4;   // 4 consecutive columns
      const int kpb = tid >> 5;         // base k-pair
      const float* srcBase = W + (size_t)kb * D_MODEL + n0 + n4;
#pragma unroll
      for (int j = 0; j < 4; ++j) {
        const int kp = kpb + 4 * j;     // k-pair 0..15
        float4 f0 = *(const float4*)(srcBase + (size_t)(2 * kp)     * D_MODEL);
        float4 f1 = *(const float4*)(srcBase + (size_t)(2 * kp + 1) * D_MODEL);
        *(uint32_t*)&sB[(n4 + 0) * 32 + 2 * kp] = f2bf_pk(f0.x, f1.x);
        *(uint32_t*)&sB[(n4 + 1) * 32 + 2 * kp] = f2bf_pk(f0.y, f1.y);
        *(uint32_t*)&sB[(n4 + 2) * 32 + 2 * kp] = f2bf_pk(f0.z, f1.z);
        *(uint32_t*)&sB[(n4 + 3) * 32 + 2 * kp] = f2bf_pk(f0.w, f1.w);
      }
    }
    __syncthreads();

    FragBF a[4], b[4];
#pragma unroll
    for (int i = 0; i < 4; ++i) {          // A 16x32: lane row, two 8-elem K chunks
      const int row = wr * 64 + i * 16 + l16;
      const int off = half * 8;
      a[i].q[0] = *(const uint4*)&sA[row * 32 + off];
      a[i].q[1] = *(const uint4*)&sA[row * 32 + off + 16];
    }
#pragma unroll
    for (int j = 0; j < 4; ++j) {          // B 32x16: lane col, two 8-elem K chunks
      const int col = wc * 64 + j * 16 + l16;
      const int off = half * 8;
      b[j].q[0] = *(const uint4*)&sB[col * 32 + off];
      b[j].q[1] = *(const uint4*)&sB[col * 32 + off + 16];
    }
#pragma unroll
    for (int i = 0; i < 4; ++i)
#pragma unroll
      for (int j = 0; j < 4; ++j)
        acc[i][j] = __builtin_amdgcn_wmma_f32_16x16x32_bf16(
            false, a[i].v, false, b[j].v, (short)0, acc[i][j], false, false);
  }

  // ---- writeout bf16 (uniform branch hoisted out of the store loops) ----
  if (wsel == 2) {
    // V transposed: Vt[d][m]
#pragma unroll
    for (int i = 0; i < 4; ++i)
#pragma unroll
      for (int j = 0; j < 4; ++j)
#pragma unroll
        for (int v = 0; v < 8; ++v) {
          const int gm = m0 + wr * 64 + i * 16 + v + 8 * half;
          const int gn = n0 + wc * 64 + j * 16 + l16;
          Vt[(size_t)gn * MTOT + gm] = f2bf(acc[i][j][v]);
        }
  } else {
    uint16_t* dst = (wsel == 0) ? Qo : Ko;
#pragma unroll
    for (int i = 0; i < 4; ++i)
#pragma unroll
      for (int j = 0; j < 4; ++j)
#pragma unroll
        for (int v = 0; v < 8; ++v) {
          const int gm = m0 + wr * 64 + i * 16 + v + 8 * half;
          const int gn = n0 + wc * 64 + j * 16 + l16;
          dst[(size_t)gm * D_MODEL + gn] = f2bf(acc[i][j][v]);
        }
  }
}

// ---------------------------------------------------------------------------
// Kernel 2: flash attention. Block = 256 thr (8 waves), q-tile 32, kv-tile 64.
// Wave w: S-tile (qs=w>>2, ks=w&3); O accum (qa=w&1, d-slice=w>>1 of 128).
// All LDS staging goes through the async-to-LDS path when available.
// ---------------------------------------------------------------------------
__global__ __launch_bounds__(256)
void attn_kernel(const uint16_t* __restrict__ Q,
                 const uint16_t* __restrict__ K,
                 const uint16_t* __restrict__ Vt,
                 float* __restrict__ out) {
  extern __shared__ char smem_raw[];
  uint16_t* sQ = (uint16_t*)smem_raw;        // [32][1024]  64 KB
  uint16_t* sK = sQ + 32 * 1024;             // [64][128]   16 KB
  uint16_t* sV = sK + 64 * 128;              // [512][64]   64 KB (V^T chunk)
  uint16_t* sP = sV + 512 * 64;              // [32][64]     4 KB
  float* sS     = (float*)(sP + 32 * 64);    // [32][64]     8 KB
  float* sAlpha = sS + 32 * 64;              // [32]
  float* sL     = sAlpha + 32;               // [32]

  const int tid  = threadIdx.x;
  const int lane = tid & 31;
  const int w    = tid >> 5;
  const int half = lane >> 4;
  const int l16  = lane & 15;

  const int qt = blockIdx.x;                 // 0..63
  const int b  = blockIdx.y;                 // 0..7
  const int qbase = qt * 32;
  const size_t mrow0 = (size_t)b * SEQ + qbase;

  const int qs  = w >> 2;                    // S-phase: q subtile 0..1
  const int ks  = w & 3;                     // S-phase: kv subtile 0..3
  const int qa  = w & 1;                     // PV/accum: q subtile
  const int dsl = w >> 1;                    // PV/accum: 128-wide d slice 0..3

  // ---- stage Q tile once: [32][1024] bf16 ----
  {
    const int r  = tid >> 3;
    const int c8 = (tid & 7) * 8;
    const uint16_t* src = Q + (mrow0 + r) * D_MODEL;
#pragma unroll
    for (int j = 0; j < 16; ++j)
      cp16(&sQ[r * 1024 + c8 + j * 64], src + c8 + j * 64);
    async_join();
  }

  v8f acc[2][8];
#pragma unroll
  for (int c = 0; c < 2; ++c)
#pragma unroll
    for (int dt = 0; dt < 8; ++dt)
#pragma unroll
      for (int v = 0; v < 8; ++v) acc[c][dt][v] = 0.0f;

  float m_prev = -3.0e38f;   // live in wave 0 lanes only
  float l_run  = 0.0f;

  const int nkv = (qt >> 1) + 1;             // causal bound, kv tiles of 64
  for (int kvt = 0; kvt < nkv; ++kvt) {
    const size_t kvm0 = (size_t)b * SEQ + (size_t)kvt * 64;

    // Prefetch next causal kv-tile of K into cache (global_prefetch_b8).
    if (kvt + 1 < nkv)
      __builtin_prefetch(K + (kvm0 + 64 + (tid >> 2)) * D_MODEL + (tid & 3) * 256, 0, 1);

    // ---- S = Q K^T  (each wave: one 16x16 tile, d streamed 128 at a time) --
    v8f sacc;
#pragma unroll
    for (int v = 0; v < 8; ++v) sacc[v] = 0.0f;

    for (int dc = 0; dc < 8; ++dc) {
      __syncthreads();
      {  // stage sK [64][128] bf16
        const int r  = tid >> 2;
        const int c8 = (tid & 3) * 8;
        const uint16_t* src = K + (kvm0 + r) * D_MODEL + dc * 128;
#pragma unroll
        for (int j = 0; j < 4; ++j)
          cp16(&sK[r * 128 + c8 + j * 32], src + c8 + j * 32);
        async_join();
      }
      __syncthreads();
#pragma unroll
      for (int ds = 0; ds < 4; ++ds) {
        FragBF af, bf;
        const int aoff = (qs * 16 + l16) * 1024 + dc * 128 + ds * 32 + half * 8;
        af.q[0] = *(const uint4*)&sQ[aoff];
        af.q[1] = *(const uint4*)&sQ[aoff + 16];
        const int boff = (ks * 16 + l16) * 128 + ds * 32 + half * 8;
        bf.q[0] = *(const uint4*)&sK[boff];
        bf.q[1] = *(const uint4*)&sK[boff + 16];
        sacc = __builtin_amdgcn_wmma_f32_16x16x32_bf16(
            false, af.v, false, bf.v, (short)0, sacc, false, false);
      }
    }

    // ---- scale + causal mask + publish S ----
#pragma unroll
    for (int v = 0; v < 8; ++v) {
      const int qg = qbase + qs * 16 + v + 8 * half;
      const int kg = kvt * 64 + ks * 16 + l16;
      float sv = sacc[v] * 0.03125f;         // 1/sqrt(1024)
      if (kg > qg) sv = -3.0e38f;
      sS[(qs * 16 + v + 8 * half) * 64 + ks * 16 + l16] = sv;
    }
    __syncthreads();

    // ---- online softmax row update (wave 0, lane = row) ----
    if (tid < 32) {
      const int r = tid;
      float rm = -3.0e38f;
#pragma unroll
      for (int j = 0; j < 16; ++j) {
        float4 f = *(const float4*)&sS[r * 64 + 4 * j];
        rm = fmaxf(rm, fmaxf(fmaxf(f.x, f.y), fmaxf(f.z, f.w)));
      }
      const float mn    = fmaxf(m_prev, rm);
      const float alpha = __expf(m_prev - mn);
      float ssum = 0.0f;
#pragma unroll
      for (int j = 0; j < 16; ++j) {
        float4 f = *(const float4*)&sS[r * 64 + 4 * j];
        const float p0 = __expf(f.x - mn), p1 = __expf(f.y - mn);
        const float p2 = __expf(f.z - mn), p3 = __expf(f.w - mn);
        ssum += (p0 + p1) + (p2 + p3);
        uint2 pk;
        pk.x = f2bf_pk(p0, p1);
        pk.y = f2bf_pk(p2, p3);
        *(uint2*)&sP[r * 64 + 4 * j] = pk;
      }
      l_run  = l_run * alpha + ssum;
      m_prev = mn;
      sAlpha[r] = alpha;
    }
    __syncthreads();

    // ---- rescale O accumulators by alpha ----
    float av[8];
#pragma unroll
    for (int v = 0; v < 8; ++v) av[v] = sAlpha[qa * 16 + v + 8 * half];
#pragma unroll
    for (int c = 0; c < 2; ++c)
#pragma unroll
      for (int dt = 0; dt < 8; ++dt)
#pragma unroll
        for (int v = 0; v < 8; ++v) acc[c][dt][v] *= av[v];

    // ---- O += P V : stream V^T in two 512-d chunks ----
#pragma unroll
    for (int c = 0; c < 2; ++c) {
      __syncthreads();
      {  // stage sV [512 d][64 kv] from V^T in ws (row = d, contiguous kv)
#pragma unroll
        for (int pass = 0; pass < 16; ++pass) {
          const int r  = pass * 32 + (tid >> 3);
          const int c8 = (tid & 7) * 8;
          cp16(&sV[r * 64 + c8],
               Vt + (size_t)(c * 512 + r) * MTOT + kvm0 + c8);
        }
        async_join();
      }
      __syncthreads();
#pragma unroll
      for (int kk = 0; kk < 2; ++kk) {       // kv reduction: 2 steps of 32
        FragBF pf;
        const int poff = (qa * 16 + l16) * 64 + kk * 32 + half * 8;
        pf.q[0] = *(const uint4*)&sP[poff];
        pf.q[1] = *(const uint4*)&sP[poff + 16];
#pragma unroll
        for (int dt = 0; dt < 8; ++dt) {
          FragBF vf;
          const int voff = (dsl * 128 + dt * 16 + l16) * 64 + kk * 32 + half * 8;
          vf.q[0] = *(const uint4*)&sV[voff];
          vf.q[1] = *(const uint4*)&sV[voff + 16];
          acc[c][dt] = __builtin_amdgcn_wmma_f32_16x16x32_bf16(
              false, pf.v, false, vf.v, (short)0, acc[c][dt], false, false);
        }
      }
    }
  }

  // ---- finalize: O /= l, write f32 ----
  if (tid < 32) sL[tid] = l_run;
  __syncthreads();
  float linv[8];
#pragma unroll
  for (int v = 0; v < 8; ++v) linv[v] = 1.0f / sL[qa * 16 + v + 8 * half];

#pragma unroll
  for (int c = 0; c < 2; ++c)
#pragma unroll
    for (int dt = 0; dt < 8; ++dt)
#pragma unroll
      for (int v = 0; v < 8; ++v) {
        const size_t row  = mrow0 + qa * 16 + v + 8 * half;
        const int    dcol = c * 512 + dsl * 128 + dt * 16 + l16;
        out[row * D_MODEL + dcol] = acc[c][dt][v] * linv[v];
      }
}

// ---------------------------------------------------------------------------
extern "C" void kernel_launch(void* const* d_in, const int* in_sizes, int n_in,
                              void* d_out, int out_size, void* d_ws, size_t ws_size,
                              hipStream_t stream) {
  const float* x  = (const float*)d_in[0];
  const float* Wq = (const float*)d_in[1];
  const float* Wk = (const float*)d_in[2];
  const float* Wv = (const float*)d_in[3];
  float* out = (float*)d_out;

  // Workspace layout (bf16): Q [16384][1024], K [16384][1024], V^T [1024][16384]
  uint16_t* Qw = (uint16_t*)d_ws;
  uint16_t* Kw = Qw + (size_t)MTOT * D_MODEL;
  uint16_t* Vt = Kw + (size_t)MTOT * D_MODEL;   // total ~100.7 MB

  dim3 g1(128, 24);
  qkv_gemm_kernel<<<g1, 128, 0, stream>>>(x, Wq, Wk, Wv, Qw, Kw, Vt);

  const size_t smem = (size_t)(32 * 1024 + 64 * 128 + 512 * 64 + 32 * 64) * sizeof(uint16_t)
                    + (size_t)(32 * 64 + 64) * sizeof(float);   // 160000 B
  dim3 g2(64, 8);
  attn_kernel<<<g2, 256, smem, stream>>>(Qw, Kw, Vt, out);
}
